// FFM_66348654789146
// MI455X (gfx1250) — matrix operand order repaired and checked by reading, not compiled, not run on hardware
//
#include <hip/hip_runtime.h>
#include <hip/hip_bf16.h>
#include <math.h>

// ---------------------------------------------------------------------------
// FFM (fast-and-forgetful memory) layer for MI455X / gfx1250.
// T=8192, IN=512, D=64, C=16, OUT=512.
// Pipeline:
//   1) f32->f16 conversion of x and all weight matrices (workspace)
//   2) WMMA GEMM:  gated = (x@W_pre^T+b) * sigmoid(x@W_gin^T+b)   [T,64] f32
//   3) chunked parallel complex scan (64 chunks x 128 steps) -> z_in [T,2048] f16
//   4) fused WMMA GEMM: z = z_in@W_mix^T, gout = sigmoid(x@W_gout^T),
//      skip = x@W_skip^T, y=z*gout, LayerNorm(y), out = ln + skip*(1-gout)
// ---------------------------------------------------------------------------

#define T_LEN 8192
#define IN_F  512
#define D_F   64
#define C_F   16
#define OUT_F 512
#define DC    (D_F * C_F)       // 1024
#define ZF    (2 * DC)          // 2048
#define NCH   64
#define CHL   128               // NCH*CHL == T_LEN
#define EPSLN 1e-5f

typedef _Float16 h8  __attribute__((ext_vector_type(8)));
typedef _Float16 h16 __attribute__((ext_vector_type(16)));
typedef float    f8  __attribute__((ext_vector_type(8)));

// ---- WMMA fragment loaders (CDNA5 wave32 16x16x32 f16 layouts) -------------
// A (16x32, MxK): lane m=lane&15, half=lane>>4; VGPR0-3 hold K = hb+0..7,
// VGPR4-7 hold K = hb+16..23 where hb = half*8.
__device__ __forceinline__ h16 load_afrag(const _Float16* row, int kb, int hh) {
    const h8 lo = *(const h8*)(row + kb + hh * 8);
    const h8 hi = *(const h8*)(row + kb + hh * 8 + 16);
    h16 r;
#pragma unroll
    for (int i = 0; i < 8; ++i) { r[i] = lo[i]; r[i + 8] = hi[i]; }
    return r;
}
// B (32x16, KxN): lane n=lane&15; elements 0-7 hold K = hb+0..7,
// elements 8-15 hold K = hb+16..23 where hb = half*8 (same as A).
__device__ __forceinline__ h16 load_bfrag(const _Float16* row, int kb, int hh) {
    const h8 lo = *(const h8*)(row + kb + hh * 8);
    const h8 hi = *(const h8*)(row + kb + hh * 8 + 16);
    h16 r;
#pragma unroll
    for (int i = 0; i < 8; ++i) { r[i] = lo[i]; r[i + 8] = hi[i]; }
    return r;
}

__device__ __forceinline__ f8 wmma16(h16 a, h16 b, f8 c) {
    return __builtin_amdgcn_wmma_f32_16x16x32_f16(false, a, false, b,
                                                  (short)0, c, false, false);
}

__device__ __forceinline__ float sigmoidf_(float t) {
    return 1.0f / (1.0f + __expf(-t));
}

// ---- 1) f32 -> f16 conversion ---------------------------------------------
__global__ void k_cvt_f16(const float* __restrict__ in, _Float16* __restrict__ out, int n) {
    int i = blockIdx.x * blockDim.x + threadIdx.x;
    if (i < n) out[i] = (_Float16)in[i];
}

// ---- 2) gated = (x@W_pre^T + b_pre) * sigmoid(x@W_gin^T + b_gin) ----------
// grid: 512 blocks (16 rows each), block: 128 threads (4 waves, one col-tile each)
__global__ void k_pre_gate(const _Float16* __restrict__ xh,
                           const _Float16* __restrict__ wpre,
                           const _Float16* __restrict__ wgin,
                           const float* __restrict__ bpre,
                           const float* __restrict__ bgin,
                           float* __restrict__ gated) {
    const int tid  = threadIdx.x;
    const int wave = tid >> 5;
    const int lane = tid & 31;
    const int hh   = lane >> 4;
    const int rb   = blockIdx.x * 16;
    const int dcol = wave * 16 + (lane & 15);

    const _Float16* arow = xh   + (size_t)(rb + (lane & 15)) * IN_F;
    const _Float16* bpr  = wpre + (size_t)dcol * IN_F;
    const _Float16* bgr  = wgin + (size_t)dcol * IN_F;

    f8 accP = {}; f8 accG = {};
#pragma unroll 4
    for (int kk = 0; kk < IN_F / 32; ++kk) {
        const int kb = kk * 32;
        h16 a  = load_afrag(arow, kb, hh);
        h16 b0 = load_bfrag(bpr,  kb, hh);
        h16 b1 = load_bfrag(bgr,  kb, hh);
        accP = wmma16(a, b0, accP);
        accG = wmma16(a, b1, accG);
    }
    const float bp = bpre[dcol];
    const float bg = bgin[dcol];
#pragma unroll
    for (int j = 0; j < 8; ++j) {
        const int m = j + hh * 8;
        const float p = accP[j] + bp;
        const float g = sigmoidf_(accG[j] + bg);
        gated[(size_t)(rb + m) * D_F + dcol] = p * g;
    }
}

// ---- 3a) per-chunk local scan (state-in = 0), emit chunk summaries --------
__global__ void k_scan_local(const float* __restrict__ gated,
                             const unsigned char* __restrict__ start,
                             const float* __restrict__ av,
                             const float* __restrict__ bv,
                             float* __restrict__ chunkEnd,   // [NCH][DC][2]
                             int* __restrict__ anyStart) {   // [NCH]
    __shared__ float lg[CHL * D_F];
    __shared__ unsigned char lst[CHL];
    const int tid = threadIdx.x;
    const int ch  = blockIdx.x;
    const int t0  = ch * CHL;

    const float4* gg = (const float4*)(gated + (size_t)t0 * D_F);
    float4* lgv = (float4*)lg;
#pragma unroll
    for (int i = tid; i < CHL * D_F / 4; i += 1024) lgv[i] = gg[i];
    if (tid < CHL) lst[tid] = start[t0 + tid];
    __syncthreads();

    const int d = tid >> 4, c = tid & 15;
    const float ad  = fabsf(av[d]);
    const float bc  = bv[c];
    const float mag = __expf(-ad);
    const float er  = mag * __cosf(bc);
    const float ei  = mag * __sinf(bc);

    float sr = 0.f, si = 0.f;
    int any = 0;
    for (int t = 0; t < CHL; ++t) {
        const int st = lst[t];
        any |= st;
        if (st) { sr = 0.f; si = 0.f; }
        const float g  = lg[t * D_F + d];
        const float nr = sr * er - si * ei + g;
        const float ni = sr * ei + si * er;
        sr = nr; si = ni;
    }
    chunkEnd[(size_t)(ch * DC + tid) * 2 + 0] = sr;
    chunkEnd[(size_t)(ch * DC + tid) * 2 + 1] = si;
    if (tid == 0) anyStart[ch] = any;
}

// ---- 3b) sequential scan over 64 chunk summaries --------------------------
__global__ void k_scan_chunks(const float* __restrict__ st_re,
                              const float* __restrict__ st_im,
                              const float* __restrict__ av,
                              const float* __restrict__ bv,
                              const float* __restrict__ chunkEnd,
                              const int* __restrict__ anyStart,
                              float* __restrict__ chunkIn,   // [NCH][DC][2]
                              float* __restrict__ stateOut)  // d_out tail, interleaved
{
    const int tid = threadIdx.x;
    const int d = tid >> 4, c = tid & 15;
    const float ad = fabsf(av[d]);
    const float bc = bv[c];
    const float mag = __expf(-(float)CHL * ad);
    const float Fr = mag * cosf((float)CHL * bc);
    const float Fi = mag * sinf((float)CHL * bc);

    float sr = st_re[tid], si = st_im[tid];
    for (int k = 0; k < NCH; ++k) {
        chunkIn[(size_t)(k * DC + tid) * 2 + 0] = sr;
        chunkIn[(size_t)(k * DC + tid) * 2 + 1] = si;
        const int  rz = anyStart[k];
        const float fr = rz ? 0.f : Fr;
        const float fi = rz ? 0.f : Fi;
        const float cer = chunkEnd[(size_t)(k * DC + tid) * 2 + 0];
        const float cei = chunkEnd[(size_t)(k * DC + tid) * 2 + 1];
        const float nr = sr * fr - si * fi + cer;
        const float ni = sr * fi + si * fr + cei;
        sr = nr; si = ni;
    }
    stateOut[tid * 2 + 0] = sr;   // complex64 interleaved [1,D,C]
    stateOut[tid * 2 + 1] = si;
}

// ---- 3c) replay each chunk with correct entry state, emit z_in (f16) ------
__global__ void k_scan_emit(const float* __restrict__ gated,
                            const unsigned char* __restrict__ start,
                            const float* __restrict__ av,
                            const float* __restrict__ bv,
                            const float* __restrict__ chunkIn,
                            _Float16* __restrict__ zin) {     // [T][2048]
    __shared__ float lg[CHL * D_F];
    __shared__ unsigned char lst[CHL];
    const int tid = threadIdx.x;
    const int ch  = blockIdx.x;
    const int t0  = ch * CHL;

    const float4* gg = (const float4*)(gated + (size_t)t0 * D_F);
    float4* lgv = (float4*)lg;
#pragma unroll
    for (int i = tid; i < CHL * D_F / 4; i += 1024) lgv[i] = gg[i];
    if (tid < CHL) lst[tid] = start[t0 + tid];
    __syncthreads();

    const int d = tid >> 4, c = tid & 15;
    const float ad  = fabsf(av[d]);
    const float bc  = bv[c];
    const float mag = __expf(-ad);
    const float er  = mag * __cosf(bc);
    const float ei  = mag * __sinf(bc);

    float sr = chunkIn[(size_t)(ch * DC + tid) * 2 + 0];
    float si = chunkIn[(size_t)(ch * DC + tid) * 2 + 1];
    for (int t = 0; t < CHL; ++t) {
        if (lst[t]) { sr = 0.f; si = 0.f; }
        const float g  = lg[t * D_F + d];
        const float nr = sr * er - si * ei + g;
        const float ni = sr * ei + si * er;
        sr = nr; si = ni;
        _Float16* zr = zin + (size_t)(t0 + t) * ZF;
        zr[tid]      = (_Float16)sr;   // real block  [0,1024)
        zr[DC + tid] = (_Float16)si;   // imag block  [1024,2048)
    }
}

// ---- 4) fused: z-GEMM + gout/skip GEMMs + gating + LayerNorm + residual ---
// grid: 512 blocks (16 output rows each), block: 1024 threads = 32 waves,
// wave w owns output columns [16w,16w+16). Dynamic LDS stages the 16-row
// z_in slab (64 KB) and x slab (16 KB) shared by all 32 waves.
__global__ void k_mix_fused(const _Float16* __restrict__ xh,
                            const _Float16* __restrict__ zin,
                            const _Float16* __restrict__ wmix,
                            const _Float16* __restrict__ wgout,
                            const _Float16* __restrict__ wskip,
                            const float* __restrict__ bmix,
                            const float* __restrict__ bgout,
                            const float* __restrict__ bskip,
                            float* __restrict__ out) {
    extern __shared__ char smem[];
    _Float16* ldsZ = (_Float16*)smem;                       // 16*2048 halves
    _Float16* ldsX = (_Float16*)(smem + 16 * ZF * 2);       // 16*512 halves
    float* ldsSum  = (float*)(smem + 16 * ZF * 2 + 16 * IN_F * 2);
    float* ldsSq   = ldsSum + 16;

    const int tid  = threadIdx.x;
    const int wave = tid >> 5;
    const int lane = tid & 31;
    const int hh   = lane >> 4;
    const int rb   = blockIdx.x * 16;
    const int ocol = wave * 16 + (lane & 15);

    { // cooperative staging (rows are contiguous in memory)
        const int4* gz = (const int4*)(zin + (size_t)rb * ZF);
        int4* lz = (int4*)ldsZ;
#pragma unroll
        for (int i = tid; i < 16 * ZF * 2 / 16; i += 1024) lz[i] = gz[i];
        const int4* gx = (const int4*)(xh + (size_t)rb * IN_F);
        int4* lx = (int4*)ldsX;
        lx[tid] = gx[tid];                 // exactly 1024 int4's
        if (tid < 16) { ldsSum[tid] = 0.f; ldsSq[tid] = 0.f; }
    }
    __syncthreads();

    const _Float16* arowZ = ldsZ + (size_t)(lane & 15) * ZF;
    const _Float16* arowX = ldsX + (size_t)(lane & 15) * IN_F;
    const _Float16* brM   = wmix  + (size_t)ocol * ZF;
    const _Float16* brG   = wgout + (size_t)ocol * IN_F;
    const _Float16* brS   = wskip + (size_t)ocol * IN_F;

    f8 accZ = {};
#pragma unroll 4
    for (int kk = 0; kk < ZF / 32; ++kk) {      // 64 WMMAs
        const int kb = kk * 32;
        accZ = wmma16(load_afrag(arowZ, kb, hh), load_bfrag(brM, kb, hh), accZ);
    }
    f8 accG = {}; f8 accS = {};
#pragma unroll 4
    for (int kk = 0; kk < IN_F / 32; ++kk) {    // 16+16 WMMAs
        const int kb = kk * 32;
        h16 a = load_afrag(arowX, kb, hh);
        accG = wmma16(a, load_bfrag(brG, kb, hh), accG);
        accS = wmma16(a, load_bfrag(brS, kb, hh), accS);
    }

    const float bm = bmix[ocol], bg = bgout[ocol], bs = bskip[ocol];
    float yv[8], gv[8], sv[8];
#pragma unroll
    for (int j = 0; j < 8; ++j) {
        const float z = accZ[j] + bm;
        const float g = sigmoidf_(accG[j] + bg);
        yv[j] = z * g;
        gv[j] = g;
        sv[j] = accS[j] + bs;
        // butterfly over the 16 lanes that share this row
        float sy = yv[j], sy2 = yv[j] * yv[j];
#pragma unroll
        for (int m = 1; m < 16; m <<= 1) {
            sy  += __shfl_xor(sy,  m, 32);
            sy2 += __shfl_xor(sy2, m, 32);
        }
        if ((lane & 15) == 0) {
            atomicAdd(&ldsSum[j + hh * 8], sy);
            atomicAdd(&ldsSq [j + hh * 8], sy2);
        }
    }
    __syncthreads();

    const float inv = 1.0f / (float)OUT_F;
#pragma unroll
    for (int j = 0; j < 8; ++j) {
        const int m = j + hh * 8;
        const float mu  = ldsSum[m] * inv;
        const float var = ldsSq[m] * inv - mu * mu;
        const float ln  = (yv[j] - mu) * rsqrtf(var + EPSLN);
        out[(size_t)(rb + m) * OUT_F + ocol] = ln + sv[j] * (1.0f - gv[j]);
    }
}

// ---------------------------------------------------------------------------
extern "C" void kernel_launch(void* const* d_in, const int* in_sizes, int n_in,
                              void* d_out, int out_size, void* d_ws, size_t ws_size,
                              hipStream_t stream) {
    (void)in_sizes; (void)n_in; (void)out_size; (void)ws_size;
    // inputs (setup_inputs order)
    const float* x       = (const float*)d_in[0];
    const float* st_re   = (const float*)d_in[1];
    const float* st_im   = (const float*)d_in[2];
    const unsigned char* start = (const unsigned char*)d_in[3];   // jax bool = 1 byte
    const float* av      = (const float*)d_in[4];
    const float* bv      = (const float*)d_in[5];
    const float* W_pre   = (const float*)d_in[6];
    const float* b_pre   = (const float*)d_in[7];
    const float* W_gin   = (const float*)d_in[8];
    const float* b_gin   = (const float*)d_in[9];
    const float* W_gout  = (const float*)d_in[10];
    const float* b_gout  = (const float*)d_in[11];
    const float* W_skip  = (const float*)d_in[12];
    const float* b_skip  = (const float*)d_in[13];
    const float* W_mix   = (const float*)d_in[14];
    const float* b_mix   = (const float*)d_in[15];
    float* out = (float*)d_out;

    // workspace layout (bytes, 256-aligned)
    char* ws = (char*)d_ws;
    _Float16* xh      = (_Float16*)(ws);                       // 8 MB
    _Float16* wmixh   = (_Float16*)(ws + 8388608);             // 2 MB
    _Float16* wgouth  = (_Float16*)(ws + 10485760);            // 512 KB
    _Float16* wskiph  = (_Float16*)(ws + 11010048);            // 512 KB
    _Float16* wpreh   = (_Float16*)(ws + 11534336);            // 64 KB
    _Float16* wginh   = (_Float16*)(ws + 11599872);            // 64 KB
    float*    gated   = (float*)   (ws + 11665408);            // 2 MB
    _Float16* zin     = (_Float16*)(ws + 13762560);            // 32 MB
    float*    chunkEnd= (float*)   (ws + 47316992);            // 512 KB
    float*    chunkIn = (float*)   (ws + 47841280);            // 512 KB
    int*      anyStart= (int*)     (ws + 48365568);            // 256 B

    auto cvt = [&](const float* src, _Float16* dst, int n) {
        k_cvt_f16<<<(n + 255) / 256, 256, 0, stream>>>(src, dst, n);
    };
    cvt(x,      xh,     T_LEN * IN_F);
    cvt(W_mix,  wmixh,  OUT_F * ZF);
    cvt(W_gout, wgouth, OUT_F * IN_F);
    cvt(W_skip, wskiph, OUT_F * IN_F);
    cvt(W_pre,  wpreh,  D_F * IN_F);
    cvt(W_gin,  wginh,  D_F * IN_F);

    // input GEMMs + gating
    k_pre_gate<<<T_LEN / 16, 128, 0, stream>>>(xh, wpreh, wginh, b_pre, b_gin, gated);

    // chunked parallel scan
    k_scan_local<<<NCH, DC, 0, stream>>>(gated, start, av, bv, chunkEnd, anyStart);
    k_scan_chunks<<<1, DC, 0, stream>>>(st_re, st_im, av, bv, chunkEnd, anyStart,
                                        chunkIn, out + (size_t)T_LEN * OUT_F);
    k_scan_emit<<<NCH, DC, 0, stream>>>(gated, start, av, bv, chunkIn, zin);

    // fused mix GEMM + gate/skip GEMMs + LayerNorm + residual
    const size_t smem = (size_t)16 * ZF * 2 + (size_t)16 * IN_F * 2 + 32 * sizeof(float);
    k_mix_fused<<<T_LEN / 16, 1024, smem, stream>>>(xh, zin, wmixh, wgouth, wskiph,
                                                    b_mix, b_gout, b_skip, out);
}
